// MultiHeadSelfAttention_70385924047539
// MI455X (gfx1250) — matrix operand order repaired and checked by reading, not compiled, not run on hardware
//
#include <hip/hip_runtime.h>
#include <hip/hip_bf16.h>
#include <cstdint>

// ---------------------------------------------------------------------------
// MI455X / gfx1250 (CDNA5, wave32) multi-head self-attention with RoPE.
// Matrix math: v_wmma_f32_16x16x32_bf16.
// GEMMs + attention K/V: LDS-staged, double-buffered with
// global_load_async_to_lds_b128 (ASYNCcnt) overlapping WMMA issue.
// ---------------------------------------------------------------------------

typedef __attribute__((ext_vector_type(16))) __bf16 v16bf;
typedef __attribute__((ext_vector_type(8)))  __bf16 v8bf;
typedef __attribute__((ext_vector_type(8)))  float  v8f;

#define D_MODEL   1024
#define NUM_HEADS 16
#define HEAD_DIM  64
#define SEQ       2048
#define BATCH     2
#define MROWS     (BATCH * SEQ)   // 4096 token rows

__device__ __forceinline__ v8f wmma_bf16(v16bf a, v16bf b, v8f c) {
  // (neg_a, A, neg_b, B, c_mod, C, reuse_a, reuse_b)
  return __builtin_amdgcn_wmma_f32_16x16x32_bf16(false, a, false, b, (short)0, c,
                                                 false, false);
}

__device__ __forceinline__ v16bf cat8(v8bf lo, v8bf hi) {
  return __builtin_shufflevector(lo, hi, 0, 1, 2, 3, 4, 5, 6, 7,
                                         8, 9, 10, 11, 12, 13, 14, 15);
}

// Async global -> LDS copy, 16B per lane (tracked by ASYNCcnt).
__device__ __forceinline__ void async_b128(uint32_t lds_off, const __bf16* g) {
  asm volatile("global_load_async_to_lds_b128 %0, %1, off"
               :: "v"(lds_off), "v"(g) : "memory");
}
__device__ __forceinline__ void wait_async0() {
  asm volatile("s_wait_asynccnt 0" ::: "memory");
}

// A-operand (16x32 bf16) gathers from an LDS tile. CDNA5 per-lane layout:
//   lanes 0-15 : row M=lane,    K = {k0+0..7,  k0+16..23}
//   lanes 16-31: row M=lane-16, K = {k0+8..15, k0+24..31}
template <int LD>
__device__ __forceinline__ v16bf lds_a(const __bf16 (&t)[LD == 64 ? 32 : 64][LD],
                                       int row, int k0, int half) {
  return cat8(*(const v8bf*)&t[row][k0 + half * 8],
              *(const v8bf*)&t[row][k0 + 16 + half * 8]);
}

// ---------------------------------------------------------------------------
// fp32 -> bf16 cast
// ---------------------------------------------------------------------------
__global__ void cast_f32_bf16(const float* __restrict__ in,
                              __bf16* __restrict__ out, int n) {
  int i = blockIdx.x * blockDim.x + threadIdx.x;
  if (i < n) out[i] = (__bf16)in[i];
}

// ---------------------------------------------------------------------------
// Y[M,N] = X[M,K] @ W[N,K]^T   (bf16 in, f32 out)
// Block: 256 threads = 8 waves. Block tile 128(M) x 128(N), K-step 64.
// Waves arranged 4(M) x 2(N); wave tile 32x64; 16 WMMAs per barrier.
// X/W K-slices (16KB each, double buffered = 64KB LDS) staged with
// global_load_async_to_lds_b128 while WMMAs consume the other buffer.
// ---------------------------------------------------------------------------
__global__ __launch_bounds__(256) void gemm_xwt_bf16(
    const __bf16* __restrict__ X, const __bf16* __restrict__ W,
    float* __restrict__ Y, int M, int N, int K) {
  __shared__ __bf16 Xs[2][128][64];
  __shared__ __bf16 Ws[2][128][64];

  const int tid  = threadIdx.x;
  const int lane = tid & 31;
  const int wid  = tid >> 5;
  const int l16  = lane & 15;
  const int half = lane >> 4;
  const int m0   = blockIdx.y * 128;
  const int n0   = blockIdx.x * 128;
  const int wm   = (wid & 3) * 32;   // wave M offset within tile
  const int wn   = (wid >> 2) * 64;  // wave N offset within tile

  // ---- copy mapping: 256 threads x 64B cover one 128x64 bf16 tile ----
  const int cr = tid >> 1;           // tile row 0..127
  const int cc = (tid & 1) * 32;     // halves 0 or 32
  const __bf16* gx = X + (size_t)(m0 + cr) * K + cc;
  const __bf16* gw = W + (size_t)(n0 + cr) * K + cc;
  const uint32_t lxs =
      (uint32_t)(uintptr_t)&Xs[0][0][0] + (uint32_t)(cr * 64 + cc) * 2;
  const uint32_t lws =
      (uint32_t)(uintptr_t)&Ws[0][0][0] + (uint32_t)(cr * 64 + cc) * 2;
  const uint32_t bufstride = 128 * 64 * 2;  // bytes per buffer

  // prologue: stage k-slice 0 into buffer 0
#pragma unroll
  for (int c = 0; c < 4; c++) {
    async_b128(lxs + c * 16, gx + c * 8);
    async_b128(lws + c * 16, gw + c * 8);
  }
  wait_async0();
  __syncthreads();

  v8f acc[2][4] = {};
  for (int k0 = 0; k0 < K; k0 += 64) {
    const int buf = (k0 >> 6) & 1;
    if (k0 + 64 < K) {  // stage next k-slice into the other buffer
      const uint32_t nb = (uint32_t)(buf ^ 1) * bufstride;
      const __bf16* gxn = gx + k0 + 64;
      const __bf16* gwn = gw + k0 + 64;
#pragma unroll
      for (int c = 0; c < 4; c++) {
        async_b128(lxs + nb + c * 16, gxn + c * 8);
        async_b128(lws + nb + c * 16, gwn + c * 8);
      }
    }

    // ---- consume current buffer from LDS: two K=32 sub-steps ----
    const __bf16(&xt)[128][64] = Xs[buf];
    const __bf16(&wt)[128][64] = Ws[buf];
#pragma unroll
    for (int ks = 0; ks < 64; ks += 32) {
      v16bf a0 = cat8(*(const v8bf*)&xt[wm + l16][ks + half * 8],
                      *(const v8bf*)&xt[wm + l16][ks + 16 + half * 8]);
      v16bf a1 = cat8(*(const v8bf*)&xt[wm + 16 + l16][ks + half * 8],
                      *(const v8bf*)&xt[wm + 16 + l16][ks + 16 + half * 8]);
      v16bf b[4];
#pragma unroll
      for (int j = 0; j < 4; j++)
        b[j] = cat8(*(const v8bf*)&wt[wn + j * 16 + l16][ks + half * 8],
                    *(const v8bf*)&wt[wn + j * 16 + l16][ks + 16 + half * 8]);
#pragma unroll
      for (int j = 0; j < 4; j++) {
        acc[0][j] = wmma_bf16(a0, b[j], acc[0][j]);
        acc[1][j] = wmma_bf16(a1, b[j], acc[1][j]);
      }
    }

    wait_async0();     // our async writes into buf^1 landed
    __syncthreads();   // everyone's writes visible; safe to overwrite buf next
  }

  // C layout: VGPR vg -> row +vg (lanes 0-15) / +vg+8 (lanes 16-31), col = lane%16
#pragma unroll
  for (int ti = 0; ti < 2; ti++) {
#pragma unroll
    for (int tj = 0; tj < 4; tj++) {
#pragma unroll
      for (int vg = 0; vg < 8; vg++) {
        int r = m0 + wm + ti * 16 + vg + half * 8;
        int c = n0 + wn + tj * 16 + l16;
        Y[(size_t)r * N + c] = acc[ti][tj][vg];
      }
    }
  }
}

// ---------------------------------------------------------------------------
// RoPE on Q,K; repack Q,K -> (B,H,S,hd) bf16; V -> (B,H,hd,S) bf16 (transposed
// so PV can load V as a WMMA A-operand with contiguous chunks).
// ---------------------------------------------------------------------------
__global__ void rope_pack(const float* __restrict__ Yq, const float* __restrict__ Yk,
                          const float* __restrict__ Yv, const int* __restrict__ pos,
                          __bf16* __restrict__ Qr, __bf16* __restrict__ Kr,
                          __bf16* __restrict__ Vt) {
  int idx = blockIdx.x * blockDim.x + threadIdx.x;
  if (idx >= BATCH * SEQ * NUM_HEADS * (HEAD_DIM / 2)) return;
  int p = idx & 31;            // pair index within head
  int h = (idx >> 5) & 15;
  int s = (idx >> 9) & 2047;
  int b = idx >> 20;

  float fpos = (float)pos[s];
  float inv_freq = __powf(10000.0f, -(float)(2 * p) / 64.0f);
  float sn, cs;
  __sincosf(fpos * inv_freq, &sn, &cs);

  size_t src = (size_t)(b * SEQ + s) * D_MODEL + h * HEAD_DIM + 2 * p;
  int bh = b * NUM_HEADS + h;
  size_t dst = ((size_t)bh * SEQ + s) * HEAD_DIM + 2 * p;

  float qe = Yq[src], qo = Yq[src + 1];
  Qr[dst]     = (__bf16)(qe * cs - qo * sn);
  Qr[dst + 1] = (__bf16)(qe * sn + qo * cs);
  float ke = Yk[src], ko = Yk[src + 1];
  Kr[dst]     = (__bf16)(ke * cs - ko * sn);
  Kr[dst + 1] = (__bf16)(ke * sn + ko * cs);

  float ve = Yv[src], vo = Yv[src + 1];
  size_t vdst = ((size_t)bh * HEAD_DIM + 2 * p) * SEQ + s;
  Vt[vdst]       = (__bf16)ve;
  Vt[vdst + SEQ] = (__bf16)vo;
}

// ---------------------------------------------------------------------------
// Flash attention. Block = 8 waves = 128 consecutive queries of one (b,h);
// all waves share K/V key blocks staged in LDS via async copies (8x traffic
// reduction vs per-wave global loads). Causal cutoff is wave-uniform, so
// compute is predicated per wave while stage/wait/barrier run on all threads.
// Scores computed TRANSPOSED (S^T = K @ Q^T) so lane = query column: softmax
// row stats need one shfl_xor(16), and P^T is rebuilt as a WMMA B-operand in
// registers. O^T = V^T @ P^T accumulated per 32-key block.
// ---------------------------------------------------------------------------
__global__ __launch_bounds__(256) void attn_flash(
    const __bf16* __restrict__ Qr, const __bf16* __restrict__ Kr,
    const __bf16* __restrict__ Vt, __bf16* __restrict__ AO) {
  __shared__ __bf16 Ks[2][32][64];   // 32 keys x 64 d
  __shared__ __bf16 Vs[2][64][32];   // 64 features x 32 keys (V^T slice)

  const int tid  = threadIdx.x;
  const int lane = tid & 31;
  const int wid  = tid >> 5;
  const int l16  = lane & 15;
  const int half = lane >> 4;
  const int bh    = blockIdx.x >> 4;        // 16 q-blocks per (b,h)
  const int qbase = (blockIdx.x & 15) * 128;
  const int q0 = qbase + wid * 16;
  const int q  = q0 + l16;                  // this lane's query row

  const __bf16* Qb = Qr + (size_t)bh * SEQ * HEAD_DIM;
  const __bf16* Kb = Kr + (size_t)bh * SEQ * HEAD_DIM;
  const __bf16* Vb = Vt + (size_t)bh * HEAD_DIM * SEQ;

  // ---- copy mapping for the shared K/V tiles (4KB each) ----
  const int kr = tid >> 3;            // K tile row 0..31
  const int kc = (tid & 7) * 8;       // halves
  const int vr = tid >> 2;            // V tile row 0..63
  const int vc = (tid & 3) * 8;       // halves
  const uint32_t lks =
      (uint32_t)(uintptr_t)&Ks[0][0][0] + (uint32_t)(kr * 64 + kc) * 2;
  const uint32_t lvs =
      (uint32_t)(uintptr_t)&Vs[0][0][0] + (uint32_t)(vr * 32 + vc) * 2;
  const uint32_t tstride = 32 * 64 * 2;   // 4KB per buffer (both tiles)

  // Q^T B-operands for d-blocks 0..31 / 32..63 (lane = query column)
  v16bf qop0 = cat8(*(const v8bf*)(Qb + (size_t)(q0 + l16) * HEAD_DIM + 0  + half * 8),
                    *(const v8bf*)(Qb + (size_t)(q0 + l16) * HEAD_DIM + 16 + half * 8));
  v16bf qop1 = cat8(*(const v8bf*)(Qb + (size_t)(q0 + l16) * HEAD_DIM + 32 + half * 8),
                    *(const v8bf*)(Qb + (size_t)(q0 + l16) * HEAD_DIM + 48 + half * 8));

  v8f o[4] = {v8f{}, v8f{}, v8f{}, v8f{}};  // O^T tiles: features 0-15..48-63
  float m_run = -1e30f, l_run = 0.0f;

  const int kend = q0 + 16;       // wave-uniform causal cutoff
  const int kmax = qbase + 128;   // block-uniform key range

  // prologue: stage key block 0 into buffer 0
  async_b128(lks, Kb + (size_t)kr * HEAD_DIM + kc);
  async_b128(lvs, Vb + (size_t)vr * SEQ + vc);
  wait_async0();
  __syncthreads();

  for (int kb = 0; kb < kmax; kb += 32) {
    const int buf = (kb >> 5) & 1;
    if (kb + 32 < kmax) {  // stage next key block (all threads)
      const uint32_t nb = (uint32_t)(buf ^ 1) * tstride;
      async_b128(lks + nb, Kb + (size_t)(kb + 32 + kr) * HEAD_DIM + kc);
      async_b128(lvs + nb, Vb + (size_t)vr * SEQ + kb + 32 + vc);
    }

    if (kb < kend) {  // wave-uniform predicate; no barriers inside
      const __bf16(&kt)[32][64] = Ks[buf];
      const __bf16(&vt)[64][32] = Vs[buf];
      // S^T tiles: s0 = keys kb..kb+15, s1 = keys kb+16..kb+31 (rows), q cols.
      v8f s0 = {}, s1 = {};
      s0 = wmma_bf16(lds_a<64>(kt, l16,      0,  half), qop0, s0);
      s1 = wmma_bf16(lds_a<64>(kt, 16 + l16, 0,  half), qop0, s1);
      s0 = wmma_bf16(lds_a<64>(kt, l16,      32, half), qop1, s0);
      s1 = wmma_bf16(lds_a<64>(kt, 16 + l16, 32, half), qop1, s1);

      // scale + causal mask + online softmax
      float p0[8], p1[8];
      float bmax = -1e30f;
#pragma unroll
      for (int vg = 0; vg < 8; vg++) {
        int key0 = kb + vg + half * 8;       // C row index of s0
        int key1 = key0 + 16;                // C row index of s1
        float v0 = (key0 <= q) ? s0[vg] * 0.125f : -1e30f;  // 1/sqrt(64)
        float v1 = (key1 <= q) ? s1[vg] * 0.125f : -1e30f;
        p0[vg] = v0; p1[vg] = v1;
        bmax = fmaxf(bmax, fmaxf(v0, v1));
      }
      bmax = fmaxf(bmax, __shfl_xor(bmax, 16, 32));
      float m_new = fmaxf(m_run, bmax);
      float corr  = __expf(m_run - m_new);
      float bsum  = 0.0f;
#pragma unroll
      for (int vg = 0; vg < 8; vg++) {
        p0[vg] = __expf(p0[vg] - m_new);
        p1[vg] = __expf(p1[vg] - m_new);
        bsum += p0[vg] + p1[vg];
      }
      bsum += __shfl_xor(bsum, 16, 32);
      l_run = l_run * corr + bsum;
      m_run = m_new;
      o[0] *= corr; o[1] *= corr; o[2] *= corr; o[3] *= corr;

      // Rebuild P^T (32 keys x 16 q) as WMMA B-operand in registers:
      // lanes 0-15 need keys kb+0..7,16..23 of q=lane; lanes 16-31 need kb+8..15,24..31.
      v16bf pt;
#pragma unroll
      for (int vg = 0; vg < 8; vg++) {
        float x0 = __shfl_xor(p0[vg], 16, 32);  // partner's s0 rows
        float x1 = __shfl_xor(p1[vg], 16, 32);  // partner's s1 rows
        float lo = p0[vg];                      // K elem vg    (or 8+vg)
        float hi = p1[vg];                      // K elem 16+vg (or 24+vg)
        pt[vg]     = (__bf16)lo;
        pt[8 + vg] = (__bf16)hi;
      }

      // O^T += V^T(16f x 32k) x P^T(32k x 16q), 4 feature tiles
      o[0] = wmma_bf16(lds_a<32>(vt,  0 + l16, 0, half), pt, o[0]);
      o[1] = wmma_bf16(lds_a<32>(vt, 16 + l16, 0, half), pt, o[1]);
      o[2] = wmma_bf16(lds_a<32>(vt, 32 + l16, 0, half), pt, o[2]);
      o[3] = wmma_bf16(lds_a<32>(vt, 48 + l16, 0, half), pt, o[3]);
    }

    wait_async0();     // our async writes into buf^1 landed
    __syncthreads();   // all threads; safe to overwrite buf next iteration
  }

  // normalize and write attn output back to token-major (B,S,D) bf16
  float inv_l = 1.0f / l_run;
  int b = bh >> 4, h = bh & 15;
  size_t rowbase = (size_t)(b * SEQ + q) * D_MODEL + h * HEAD_DIM + half * 8;
#pragma unroll
  for (int ft = 0; ft < 4; ft++) {
    v8bf outv;
#pragma unroll
    for (int vg = 0; vg < 8; vg++) outv[vg] = (__bf16)(o[ft][vg] * inv_l);
    *(v8bf*)(AO + rowbase + ft * 16) = outv;
  }
}

// ---------------------------------------------------------------------------
// launch
// ---------------------------------------------------------------------------
extern "C" void kernel_launch(void* const* d_in, const int* in_sizes, int n_in,
                              void* d_out, int out_size, void* d_ws, size_t ws_size,
                              hipStream_t stream) {
  (void)in_sizes; (void)n_in; (void)out_size; (void)ws_size;
  const float* x   = (const float*)d_in[0];
  const int*   pos = (const int*)  d_in[1];
  const float* WQ  = (const float*)d_in[2];
  const float* WK  = (const float*)d_in[3];
  const float* WV  = (const float*)d_in[4];
  const float* WO  = (const float*)d_in[5];
  float* out = (float*)d_out;

  char* ws = (char*)d_ws;
  const size_t MB = 1024 * 1024;
  __bf16* xb  = (__bf16*)(ws + 0);        //  8 MB  x bf16 (4096x1024)
  __bf16* wqb = (__bf16*)(ws + 8 * MB);   //  2 MB
  __bf16* wkb = (__bf16*)(ws + 10 * MB);  //  2 MB
  __bf16* wvb = (__bf16*)(ws + 12 * MB);  //  2 MB
  __bf16* wob = (__bf16*)(ws + 14 * MB);  //  2 MB
  float*  Yq  = (float*) (ws + 16 * MB);  // 16 MB  Q pre-RoPE f32
  float*  Yk  = (float*) (ws + 32 * MB);  // 16 MB
  float*  Yv  = (float*) (ws + 48 * MB);  // 16 MB
  __bf16* Qr  = (__bf16*)(ws + 64 * MB);  //  8 MB  (B,H,S,hd)
  __bf16* Kr  = (__bf16*)(ws + 72 * MB);  //  8 MB  (B,H,S,hd)
  __bf16* Vt  = (__bf16*)(ws + 80 * MB);  //  8 MB  (B,H,hd,S)
  __bf16* AO  = (__bf16*)(ws + 88 * MB);  //  8 MB  attn out (B,S,D)

  const int nx = MROWS * D_MODEL;
  const int nw = D_MODEL * D_MODEL;
  cast_f32_bf16<<<(nx + 255) / 256, 256, 0, stream>>>(x,  xb,  nx);
  cast_f32_bf16<<<(nw + 255) / 256, 256, 0, stream>>>(WQ, wqb, nw);
  cast_f32_bf16<<<(nw + 255) / 256, 256, 0, stream>>>(WK, wkb, nw);
  cast_f32_bf16<<<(nw + 255) / 256, 256, 0, stream>>>(WV, wvb, nw);
  cast_f32_bf16<<<(nw + 255) / 256, 256, 0, stream>>>(WO, wob, nw);

  dim3 gg(D_MODEL / 128, MROWS / 128);   // (8, 32)
  gemm_xwt_bf16<<<gg, 256, 0, stream>>>(xb, wqb, Yq, MROWS, D_MODEL, D_MODEL);
  gemm_xwt_bf16<<<gg, 256, 0, stream>>>(xb, wkb, Yk, MROWS, D_MODEL, D_MODEL);
  gemm_xwt_bf16<<<gg, 256, 0, stream>>>(xb, wvb, Yv, MROWS, D_MODEL, D_MODEL);

  const int nrope = BATCH * SEQ * NUM_HEADS * (HEAD_DIM / 2);
  rope_pack<<<(nrope + 255) / 256, 256, 0, stream>>>(Yq, Yk, Yv, pos, Qr, Kr, Vt);

  attn_flash<<<(BATCH * NUM_HEADS * (SEQ / 16)) / 8, 256, 0, stream>>>(Qr, Kr, Vt, AO);

  gemm_xwt_bf16<<<gg, 256, 0, stream>>>(AO, wob, out, MROWS, D_MODEL, D_MODEL);
}
